// GRUCNN_74071005987545
// MI455X (gfx1250) — matrix operand order repaired and checked by reading, not compiled, not run on hardware
//
#include <hip/hip_runtime.h>

typedef _Float16 v8h  __attribute__((ext_vector_type(8)));
typedef _Float16 v16h __attribute__((ext_vector_type(16)));
typedef float    v8f  __attribute__((ext_vector_type(8)));

#define LOG2E 1.4426950408889634f

// ---------------- weight fp32 -> f16 ----------------
__global__ void k_cvt(const float* __restrict__ in, _Float16* __restrict__ out, int n) {
    int t = blockIdx.x * blockDim.x + threadIdx.x;
    if (t < n) out[t] = (_Float16)in[t];
}

// ---------------- NCHW fp32 -> NHWC f16 (input image) ----------------
__global__ void k_nchw_to_nhwc(const float* __restrict__ in, _Float16* __restrict__ out,
                               int B, int C, int H, int W) {
    int t = blockIdx.x * blockDim.x + threadIdx.x;
    int total = B * C * H * W;
    if (t >= total) return;
    int c = t % C;
    int w = (t / C) % W;
    int h = (t / (C * W)) % H;
    int b = t / (C * W * H);
    out[t] = (_Float16)in[(((size_t)b * C + c) * H + h) * W + w];
}

// ---------------- small-K conv (Ci==3), NHWC f16, fused ReLU ----------------
__global__ void k_conv_small(const _Float16* __restrict__ in, const _Float16* __restrict__ w,
                             const float* __restrict__ bias, _Float16* __restrict__ out,
                             int P, int Ci, int Co) {
    int t = blockIdx.x * blockDim.x + threadIdx.x;
    if (t >= P * Co) return;
    int p = t / Co, co = t % Co;
    float s = bias[co];
    for (int c = 0; c < Ci; ++c)
        s += (float)in[(size_t)p * Ci + c] * (float)w[co * Ci + c];
    out[(size_t)p * Co + co] = (_Float16)fmaxf(s, 0.f);
}

// ---------------- WMMA 1x1 conv: D[P x Co] = X[P x Ci] * W^T, fused bias+ReLU ----------------
// NHWC f16 activations; weights row-major [Co x Ci] f16.
// One wave computes a 32x64 output block = 2 (pixel) x 4 (channel) 16x16 WMMA tiles:
// 8 independent accumulator chains, 8 WMMAs per K-step from 12 b128 fragment loads.
__global__ void __launch_bounds__(32)
k_conv_wmma(const _Float16* __restrict__ in, const _Float16* __restrict__ w,
            const float* __restrict__ bias, _Float16* __restrict__ out,
            int P, int Ci, int Co) {
    const int lane  = threadIdx.x;        // 0..31, wave32
    const int hi    = lane >> 4;          // 0 = lanes 0-15, 1 = lanes 16-31
    const int l     = lane & 15;
    const int ptile = blockIdx.x << 5;    // 32 pixels
    const int ctile = blockIdx.y << 6;    // 64 output channels

    // A fragment: row m = l ; lo-lanes K = 0..7 & 16..23, hi-lanes +8
    const _Float16* aBase0 = in + (size_t)(ptile + l) * Ci + hi * 8;
    const _Float16* aBase1 = aBase0 + (size_t)16 * Ci;
    // B fragment: col n = l ; lo-lanes K = 0..7 & 16..23, hi-lanes +8
    const _Float16* bBase0 = w + (size_t)(ctile + l) * Ci + hi * 8;

    v8f acc[2][4];
    for (int j = 0; j < 4; ++j) {
        float bv = bias[ctile + 16 * j + l];   // bias indexed by output channel (N = l)
        for (int i = 0; i < 8; ++i) { acc[0][j][i] = bv; acc[1][j][i] = bv; }
    }

    for (int k = 0; k < Ci; k += 32) {
        v16h A[2], B[4];
        {
            v8h a0 = *(const v8h*)(aBase0 + k);
            v8h a1 = *(const v8h*)(aBase0 + k + 16);
            A[0] = __builtin_shufflevector(a0, a1, 0,1,2,3,4,5,6,7,8,9,10,11,12,13,14,15);
            v8h a2 = *(const v8h*)(aBase1 + k);
            v8h a3 = *(const v8h*)(aBase1 + k + 16);
            A[1] = __builtin_shufflevector(a2, a3, 0,1,2,3,4,5,6,7,8,9,10,11,12,13,14,15);
        }
        for (int j = 0; j < 4; ++j) {
            const _Float16* bb = bBase0 + (size_t)16 * j * Ci + k;
            v8h b0 = *(const v8h*)(bb);
            v8h b1 = *(const v8h*)(bb + 16);
            B[j] = __builtin_shufflevector(b0, b1, 0,1,2,3,4,5,6,7,8,9,10,11,12,13,14,15);
        }
        for (int i = 0; i < 2; ++i)
            for (int j = 0; j < 4; ++j)
                acc[i][j] = __builtin_amdgcn_wmma_f32_16x16x32_f16(
                    false, A[i], false, B[j], (short)0, acc[i][j], false, false);
    }

    // D layout: VGPR r -> m = r + 8*hi, n = l
    for (int i = 0; i < 2; ++i) {
        for (int j = 0; j < 4; ++j) {
            _Float16* o = out + (size_t)(ptile + 16 * i) * Co + ctile + 16 * j + l;
            for (int r = 0; r < 8; ++r) {
                int m = r + 8 * hi;
                o[(size_t)m * Co] = (_Float16)fmaxf(acc[i][j][r], 0.f);
            }
        }
    }
}

// ---------------- 2x2 max pool, NHWC f16 ----------------
__global__ void k_pool(const _Float16* __restrict__ in, _Float16* __restrict__ out,
                       int B, int H, int W, int C) {   // H,W = input dims
    int t = blockIdx.x * blockDim.x + threadIdx.x;
    int Ho = H >> 1, Wo = W >> 1;
    int total = B * Ho * Wo * C;
    if (t >= total) return;
    int c = t % C;
    int x = (t / C) % Wo;
    int y = (t / (C * Wo)) % Ho;
    int b = t / (C * Wo * Ho);
    const _Float16* p = in + (((size_t)b * H + 2 * y) * W + 2 * x) * C + c;
    float m0 = fmaxf((float)p[0],             (float)p[C]);
    float m1 = fmaxf((float)p[(size_t)W * C], (float)p[(size_t)W * C + C]);
    out[t] = (_Float16)fmaxf(m0, m1);
}

// ---- NHWC (8,7,7,512) f16 -> precomputed x-dependent gate terms (NCHW-flat order) ----
// a0[t] = -log2e*(wi0*x+bi0+bh0) ; a1[t] = -log2e*(wi1*x+bi1+bh1) ; g2[t] = wi2*x+bi2
__global__ void k_pre(const _Float16* __restrict__ act,
                      const float* __restrict__ wi, const float* __restrict__ bi,
                      const float* __restrict__ bh,
                      float* __restrict__ a0, float* __restrict__ a1,
                      float* __restrict__ g2, int L) {
    int t = blockIdx.x * blockDim.x + threadIdx.x;
    if (t >= L) return;
    int w = t % 7;
    int h = (t / 7) % 7;
    int c = (t / 49) % 512;
    int b = t / 25088;                    // 49*512
    float x = (float)act[((((size_t)b * 7) + h) * 7 + w) * 512 + c];
    a0[t] = -LOG2E * (wi[0] * x + bi[0] + bh[0]);
    a1[t] = -LOG2E * (wi[1] * x + bi[1] + bh[1]);
    g2[t] = wi[2] * x + bi[2];
}

// ---------------- serial scalar GRU scan, ys = relu(h_t) ----------------
// sigmoid via v_exp_f32/v_rcp_f32 (2^x form), tanh via v_tanh_f32 when available.
__global__ void k_gru(const float* __restrict__ a0, const float* __restrict__ a1,
                      const float* __restrict__ g2,
                      const float* __restrict__ wh, const float* __restrict__ bh,
                      float* __restrict__ ys, int L) {
    if (threadIdx.x != 0 || blockIdx.x != 0) return;
    const float k0 = -LOG2E * wh[0];
    const float k1 = -LOG2E * wh[1];
    const float wh2 = wh[2], bh2 = bh[2];
    float h = 0.f;
    for (int t = 0; t < L; ++t) {
        __builtin_prefetch(a0 + t + 128, 0, 0);
        __builtin_prefetch(a1 + t + 128, 0, 0);
        __builtin_prefetch(g2 + t + 128, 0, 0);
        float r = __builtin_amdgcn_rcpf(
                      1.f + __builtin_amdgcn_exp2f(__builtin_fmaf(k0, h, a0[t])));
        float z = __builtin_amdgcn_rcpf(
                      1.f + __builtin_amdgcn_exp2f(__builtin_fmaf(k1, h, a1[t])));
        float narg = __builtin_fmaf(r, __builtin_fmaf(wh2, h, bh2), g2[t]);
#if __has_builtin(__builtin_amdgcn_tanhf)
        float n = __builtin_amdgcn_tanhf(narg);
#else
        float n = tanhf(narg);
#endif
        h = __builtin_fmaf(z, h - n, n);  // (1-z)*n + z*h
        ys[t] = fmaxf(h, 0.f);
    }
}

// ---------------- fc1 GEMV: acc[j] = fc1_w[j,:] . ys  (one block per row) ----------------
__global__ void k_fc1(const float* __restrict__ fw, const float* __restrict__ ys,
                      float* __restrict__ acc, int L) {
    __shared__ float red[256];
    const int j = blockIdx.x;
    const float* row = fw + (size_t)j * L;
    float s = 0.f;
    for (int t = threadIdx.x; t < L; t += 256) {
        __builtin_prefetch(row + t + 4096, 0, 0);
        s += row[t] * ys[t];
    }
    red[threadIdx.x] = s;
    __syncthreads();
    for (int o = 128; o > 0; o >>= 1) {
        if (threadIdx.x < o) red[threadIdx.x] += red[threadIdx.x + o];
        __syncthreads();
    }
    if (threadIdx.x == 0) acc[j] = red[0];
}

// ---------------- head: relu(acc+b1) -> fc2 -> softmax ----------------
__global__ void k_head(const float* __restrict__ acc, const float* __restrict__ b1,
                       const float* __restrict__ w2, const float* __restrict__ b2,
                       float* __restrict__ out) {
    if (threadIdx.x != 0 || blockIdx.x != 0) return;
    float hh[64];
    for (int j = 0; j < 64; ++j) hh[j] = fmaxf(acc[j] + b1[j], 0.f);
    float lg[3];
    float mx = -1e30f;
    for (int i = 0; i < 3; ++i) {
        float s = b2[i];
        for (int j = 0; j < 64; ++j) s += w2[i * 64 + j] * hh[j];
        lg[i] = s;
        mx = fmaxf(mx, s);
    }
    float e0 = __expf(lg[0] - mx), e1 = __expf(lg[1] - mx), e2 = __expf(lg[2] - mx);
    float inv = 1.f / (e0 + e1 + e2);
    out[0] = e0 * inv; out[1] = e1 * inv; out[2] = e2 * inv;
}

static inline int cdiv(int a, int b) { return (a + b - 1) / b; }

extern "C" void kernel_launch(void* const* d_in, const int* in_sizes, int n_in,
                              void* d_out, int out_size, void* d_ws, size_t ws_size,
                              hipStream_t stream) {
    (void)in_sizes; (void)n_in; (void)out_size; (void)ws_size;
    // input order: x, (conv w,b) x10, gru_wi, gru_wh, gru_bi, gru_bh, fc1_w, fc1_b, fc2_w, fc2_b
    const float* x = (const float*)d_in[0];
    const float* cw[10]; const float* cb[10];
    for (int i = 0; i < 10; ++i) { cw[i] = (const float*)d_in[1 + 2 * i];
                                   cb[i] = (const float*)d_in[2 + 2 * i]; }
    const float* gwi = (const float*)d_in[21];
    const float* gwh = (const float*)d_in[22];
    const float* gbi = (const float*)d_in[23];
    const float* gbh = (const float*)d_in[24];
    const float* f1w = (const float*)d_in[25];
    const float* f1b = (const float*)d_in[26];
    const float* f2w = (const float*)d_in[27];
    const float* f2b = (const float*)d_in[28];
    float* out = (float*)d_out;

    char* ws = (char*)d_ws;
    _Float16* w16  = (_Float16*)(ws);                      //  ~2.1 MB of f16 weights
    _Float16* actA = (_Float16*)(ws + ((size_t)4   << 20)); // 52 MB
    _Float16* actB = (_Float16*)(ws + ((size_t)56  << 20)); // 52 MB
    float*    a0   = (float*)   (ws + ((size_t)108 << 20));
    float*    a1   = (float*)   (ws + ((size_t)109 << 20));
    float*    g2   = (float*)   (ws + ((size_t)110 << 20));
    float*    ys   = (float*)   (ws + ((size_t)111 << 20));
    float*    acc  = (float*)   (ws + ((size_t)112 << 20));

    static const int CI[10] = {3, 3, 64, 128, 128, 256, 256, 512, 512, 512};
    static const int CO[10] = {3, 64, 128, 128, 256, 256, 512, 512, 512, 512};
    size_t woff[10], o = 0;
    for (int i = 0; i < 10; ++i) { woff[i] = o; o += ((size_t)CI[i] * CO[i] + 63) & ~(size_t)63; }

    // weights -> f16
    for (int i = 0; i < 10; ++i) {
        int n = CI[i] * CO[i];
        k_cvt<<<cdiv(n, 256), 256, 0, stream>>>(cw[i], w16 + woff[i], n);
    }
    // input NCHW fp32 -> NHWC f16
    { int n = 8 * 3 * 224 * 224;
      k_nchw_to_nhwc<<<cdiv(n, 256), 256, 0, stream>>>(x, actA, 8, 3, 224, 224); }

    const int L = 8 * 512 * 7 * 7;  // 200704
    int P;

    // block 0 (Ci=3 layers on scalar path)
    P = 8 * 224 * 224;
    k_conv_small<<<cdiv(P * 3, 256), 256, 0, stream>>>(actA, w16 + woff[0], cb[0], actB, P, 3, 3);
    k_conv_small<<<cdiv(P * 64, 256), 256, 0, stream>>>(actB, w16 + woff[1], cb[1], actA, P, 3, 64);
    k_pool<<<cdiv(8 * 112 * 112 * 64, 256), 256, 0, stream>>>(actA, actB, 8, 224, 224, 64);

    // block 1
    P = 8 * 112 * 112;
    k_conv_wmma<<<dim3(P >> 5, 128 >> 6), 32, 0, stream>>>(actB, w16 + woff[2], cb[2], actA, P, 64, 128);
    k_conv_wmma<<<dim3(P >> 5, 128 >> 6), 32, 0, stream>>>(actA, w16 + woff[3], cb[3], actB, P, 128, 128);
    k_pool<<<cdiv(8 * 56 * 56 * 128, 256), 256, 0, stream>>>(actB, actA, 8, 112, 112, 128);

    // block 2
    P = 8 * 56 * 56;
    k_conv_wmma<<<dim3(P >> 5, 256 >> 6), 32, 0, stream>>>(actA, w16 + woff[4], cb[4], actB, P, 128, 256);
    k_conv_wmma<<<dim3(P >> 5, 256 >> 6), 32, 0, stream>>>(actB, w16 + woff[5], cb[5], actA, P, 256, 256);
    k_pool<<<cdiv(8 * 28 * 28 * 256, 256), 256, 0, stream>>>(actA, actB, 8, 56, 56, 256);

    // block 3
    P = 8 * 28 * 28;
    k_conv_wmma<<<dim3(P >> 5, 512 >> 6), 32, 0, stream>>>(actB, w16 + woff[6], cb[6], actA, P, 256, 512);
    k_conv_wmma<<<dim3(P >> 5, 512 >> 6), 32, 0, stream>>>(actA, w16 + woff[7], cb[7], actB, P, 512, 512);
    k_pool<<<cdiv(8 * 14 * 14 * 512, 256), 256, 0, stream>>>(actB, actA, 8, 28, 28, 512);

    // block 4
    P = 8 * 14 * 14;
    k_conv_wmma<<<dim3(P >> 5, 512 >> 6), 32, 0, stream>>>(actA, w16 + woff[8], cb[8], actB, P, 512, 512);
    k_conv_wmma<<<dim3(P >> 5, 512 >> 6), 32, 0, stream>>>(actB, w16 + woff[9], cb[9], actA, P, 512, 512);
    k_pool<<<cdiv(8 * 7 * 7 * 512, 256), 256, 0, stream>>>(actA, actB, 8, 14, 14, 512);

    // GRU + head
    k_pre<<<cdiv(L, 256), 256, 0, stream>>>(actB, gwi, gbi, gbh, a0, a1, g2, L);
    k_gru<<<1, 1, 0, stream>>>(a0, a1, g2, gwh, gbh, ys, L);
    k_fc1<<<64, 256, 0, stream>>>(f1w, ys, acc, L);
    k_head<<<1, 1, 0, stream>>>(acc, f1b, f2w, f2b, out);
}
